// GroupFC_54820962566783
// MI455X (gfx1250) — matrix operand order repaired and checked by reading, not compiled, not run on hardware
//
#include <hip/hip_runtime.h>

// ---------------------------------------------------------------------------
// GroupFC with cosine normalization:
//   out[b,g,k] = <h[b,g,:], W[g,:,k]> / (max(|h[b,g,:]|,eps) * max(|W[g,:,k]|,eps))
// B=256, G=100, D=768, K=128  (all fp32 in / fp32 out)
//
// Strategy: norms in fp32 (exact), operands converted to bf16,
// GEMM via v_wmma_f32_16x16x32_bf16, scales applied in the epilogue.
// ---------------------------------------------------------------------------

#define BB 256
#define GG 100
#define DD 768
#define KK 128

typedef __attribute__((ext_vector_type(16))) __bf16 v16bf;
typedef __attribute__((ext_vector_type(8)))  float  v8f;

union ABFrag {
    uint4 q[2];
    v16bf v;
};

__device__ __forceinline__ unsigned short f2bf(float f) {
    unsigned u = __float_as_uint(f);
    unsigned r = u + 0x7FFFu + ((u >> 16) & 1u);   // round-to-nearest-even
    return (unsigned short)(r >> 16);
}

// ---------------------------------------------------------------------------
// Kernel 1: per-row (b,g) inverse norm of h -> ih, and bf16 conversion of h.
// One wave (32 lanes) per row of 768 floats. 8 waves / block.
// ---------------------------------------------------------------------------
__global__ void __launch_bounds__(256) hprep_kernel(const float* __restrict__ h,
                                                    unsigned short* __restrict__ hb,
                                                    float* __restrict__ ih) {
    const int lane = threadIdx.x & 31;
    const int wave = threadIdx.x >> 5;
    const int row  = blockIdx.x * 8 + wave;              // 0 .. B*G-1 (25600)

    const float* src = h + (size_t)row * DD;
    unsigned short* dst = hb + (size_t)row * DD;

    float ss = 0.0f;
#pragma unroll
    for (int i = 0; i < 6; ++i) {
        const int d = i * 128 + lane * 4;
        float4 v = *(const float4*)(src + d);
        ss += v.x * v.x + v.y * v.y + v.z * v.z + v.w * v.w;
        uint2 p;
        p.x = (unsigned)f2bf(v.x) | ((unsigned)f2bf(v.y) << 16);
        p.y = (unsigned)f2bf(v.z) | ((unsigned)f2bf(v.w) << 16);
        *(uint2*)(dst + d) = p;
    }
#pragma unroll
    for (int off = 16; off >= 1; off >>= 1)
        ss += __shfl_xor(ss, off, 32);
    if (lane == 0)
        ih[row] = 1.0f / fmaxf(sqrtf(ss), 1e-12f);
}

// ---------------------------------------------------------------------------
// Kernel 2: per-column (g,k) inverse norm of W. One thread per column; lanes
// with consecutive k read contiguous 128B lines each d-step.
// ---------------------------------------------------------------------------
__global__ void __launch_bounds__(256) wnorm_kernel(const float* __restrict__ W,
                                                    float* __restrict__ iw) {
    const int t = blockIdx.x * 256 + threadIdx.x;        // 0 .. G*K-1 (12800)
    const int g = t >> 7;
    const int k = t & (KK - 1);
    const float* col = W + (size_t)g * DD * KK + k;
    float ss = 0.0f;
    for (int d = 0; d < DD; ++d) {
        float f = col[(size_t)d * KK];
        ss += f * f;
    }
    iw[t] = 1.0f / fmaxf(sqrtf(ss), 1e-12f);
}

// ---------------------------------------------------------------------------
// Kernel 3: W [G,D,K] fp32 -> Wt [G,K,D] bf16 via 32x32 LDS tile transpose.
// Makes the WMMA B-fragment (16 contiguous K-dim values per lane) two
// contiguous b128 loads in the GEMM.
// ---------------------------------------------------------------------------
__global__ void __launch_bounds__(256) wtrans_kernel(const float* __restrict__ W,
                                                     unsigned short* __restrict__ wt) {
    __shared__ unsigned short lds[32][33];

    const int b   = blockIdx.x;          // G * (D/32) * (K/32) = 100*24*4 = 9600
    const int g   = b / 96;
    const int rem = b % 96;
    const int dt  = rem >> 2;            // 0..23
    const int kt  = rem & 3;             // 0..3

    const int ki = threadIdx.x & 31;
    const int dj = (threadIdx.x >> 5) * 4;

    const float* base = W + ((size_t)g * DD + dt * 32) * KK + kt * 32;
#pragma unroll
    for (int j = 0; j < 4; ++j) {
        float f = base[(size_t)(dj + j) * KK + ki];   // coalesced in k across lanes
        lds[ki][dj + j] = f2bf(f);
    }
    __syncthreads();

    const int ko = threadIdx.x >> 3;          // 0..31
    const int dq = (threadIdx.x & 7) * 4;     // 0..28
    uint2 p;
    p.x = (unsigned)lds[ko][dq]     | ((unsigned)lds[ko][dq + 1] << 16);
    p.y = (unsigned)lds[ko][dq + 2] | ((unsigned)lds[ko][dq + 3] << 16);
    *(uint2*)(wt + ((size_t)g * KK + kt * 32 + ko) * DD + dt * 32 + dq) = p;
}

// ---------------------------------------------------------------------------
// Kernel 4: grouped GEMM via v_wmma_f32_16x16x32_bf16.
// Each wave computes a 16(M) x 64(N) output tile: 4 accumulators, A fragment
// reused across the 4 WMMAs per K-step. 32 waves per group, 8 waves / block,
// 400 blocks total. EXEC is all-1s everywhere (exact grid).
//
// A fragment: lane l<16 holds row m0+l, K = d0+0..15 contiguous;
// lane l>=16 holds row m0+l-16, K = d0+16..31 (same k order as B).
// B fragment: lane l<16 holds column n (=lane), K = d0+0..15 contiguous;
// lanes 16..31 hold K = d0+16..31.
// ---------------------------------------------------------------------------
__global__ void __launch_bounds__(256) gemm_kernel(const unsigned short* __restrict__ hb,
                                                   const unsigned short* __restrict__ wt,
                                                   const float* __restrict__ ih,
                                                   const float* __restrict__ iw,
                                                   float* __restrict__ out) {
    const int lane = threadIdx.x & 31;
    const int wave = threadIdx.x >> 5;
    const int w    = blockIdx.x * 8 + wave;      // 0..3199
    const int g    = w >> 5;                     // 0..99
    const int r    = w & 31;
    const int m0   = (r & 15) << 4;              // 0,16,...,240
    const int n0   = (r >> 4) << 6;              // 0 or 64

    const int la16  = lane & 15;
    const int kbA   = (lane >> 4) ? 16 : 0;      // A fragment K sub-offset
    const int kbB   = (lane >> 4) ? 16 : 0;      // B fragment K sub-offset

    const unsigned short* arow = hb + ((size_t)(m0 + la16) * GG + g) * DD + kbA;
    const unsigned short* b0r  = wt + ((size_t)g * KK + (n0 +  0 + la16)) * DD + kbB;
    const unsigned short* b1r  = wt + ((size_t)g * KK + (n0 + 16 + la16)) * DD + kbB;
    const unsigned short* b2r  = wt + ((size_t)g * KK + (n0 + 32 + la16)) * DD + kbB;
    const unsigned short* b3r  = wt + ((size_t)g * KK + (n0 + 48 + la16)) * DD + kbB;

    v8f c0 = {}, c1 = {}, c2 = {}, c3 = {};

#pragma unroll 4
    for (int d0 = 0; d0 < DD; d0 += 32) {
        ABFrag a, b0, b1, b2, b3;
        a.q[0]  = *(const uint4*)(arow + d0);        // K = kbA + 0..7
        a.q[1]  = *(const uint4*)(arow + d0 + 8);    // K = kbA + 8..15
        b0.q[0] = *(const uint4*)(b0r + d0);
        b0.q[1] = *(const uint4*)(b0r + d0 + 8);
        b1.q[0] = *(const uint4*)(b1r + d0);
        b1.q[1] = *(const uint4*)(b1r + d0 + 8);
        b2.q[0] = *(const uint4*)(b2r + d0);
        b2.q[1] = *(const uint4*)(b2r + d0 + 8);
        b3.q[0] = *(const uint4*)(b3r + d0);
        b3.q[1] = *(const uint4*)(b3r + d0 + 8);

        c0 = __builtin_amdgcn_wmma_f32_16x16x32_bf16(false, a.v, false, b0.v,
                                                     (short)0, c0, false, false);
        c1 = __builtin_amdgcn_wmma_f32_16x16x32_bf16(false, a.v, false, b1.v,
                                                     (short)0, c1, false, false);
        c2 = __builtin_amdgcn_wmma_f32_16x16x32_bf16(false, a.v, false, b2.v,
                                                     (short)0, c2, false, false);
        c3 = __builtin_amdgcn_wmma_f32_16x16x32_bf16(false, a.v, false, b3.v,
                                                     (short)0, c3, false, false);
    }

    // Epilogue: scale by inverse norms and store fp32.
    // C layout: VGPR j -> row m0 + j + (lane>=16 ? 8 : 0), column = lane&15.
    const int madd = (lane >> 4) * 8;
    float sh[8];
#pragma unroll
    for (int j = 0; j < 8; ++j)
        sh[j] = ih[(size_t)(m0 + j + madd) * GG + g];

    const float* iwg = iw + (size_t)g * KK + n0 + la16;
    const float sw0 = iwg[0];
    const float sw1 = iwg[16];
    const float sw2 = iwg[32];
    const float sw3 = iwg[48];

#pragma unroll
    for (int j = 0; j < 8; ++j) {
        const size_t o = ((size_t)(m0 + j + madd) * GG + g) * KK + n0 + la16;
        out[o +  0] = c0[j] * sh[j] * sw0;
        out[o + 16] = c1[j] * sh[j] * sw1;
        out[o + 32] = c2[j] * sh[j] * sw2;
        out[o + 48] = c3[j] * sh[j] * sw3;
    }
}

// ---------------------------------------------------------------------------
// Host launcher
// ---------------------------------------------------------------------------
extern "C" void kernel_launch(void* const* d_in, const int* in_sizes, int n_in,
                              void* d_out, int out_size, void* d_ws, size_t ws_size,
                              hipStream_t stream) {
    const float* h = (const float*)d_in[0];   // [B,G,D] fp32
    const float* W = (const float*)d_in[1];   // [G,D,K] fp32
    float* out     = (float*)d_out;           // [B,G,K] fp32

    // Workspace layout (~59.1 MB):
    //   hb : B*G*D bf16  = 39,321,600 B
    //   wt : G*K*D bf16  = 19,660,800 B
    //   ih : B*G   fp32  =    102,400 B
    //   iw : G*K   fp32  =     51,200 B
    const size_t HB_BYTES = (size_t)BB * GG * DD * 2;
    const size_t WT_BYTES = (size_t)GG * KK * DD * 2;
    char* ws = (char*)d_ws;
    unsigned short* hb = (unsigned short*)ws;
    unsigned short* wt = (unsigned short*)(ws + HB_BYTES);
    float* ih = (float*)(ws + HB_BYTES + WT_BYTES);
    float* iw = ih + (size_t)BB * GG;

    hprep_kernel <<<(BB * GG) / 8,          256, 0, stream>>>(h, hb, ih);
    wnorm_kernel <<<(GG * KK) / 256,        256, 0, stream>>>(W, iw);
    wtrans_kernel<<<GG * (DD / 32) * (KK / 32), 256, 0, stream>>>(W, wt);
    gemm_kernel  <<<(GG * 32) / 8,          256, 0, stream>>>(hb, wt, ih, iw, out);
}
